// GaussianRenderer_90185723282029
// MI455X (gfx1250) — matrix operand order repaired and checked by reading, not compiled, not run on hardware
//
#include <hip/hip_runtime.h>
#include <math.h>

typedef __attribute__((ext_vector_type(16))) _Float16 v16h;
typedef __attribute__((ext_vector_type(8)))  _Float16 v8h;
typedef __attribute__((ext_vector_type(8)))  float    v8f;

#define WAVES_PER_BLOCK 8
#define MAXN 1024

// -----------------------------------------------------------------------------
// Layout prep kernels (one-shot, memory-bound, negligible at 23.3 TB/s)
// -----------------------------------------------------------------------------
__global__ void k_cvt_f16(const float* __restrict__ src, _Float16* __restrict__ dst, int n)
{
    int i = blockIdx.x * blockDim.x + threadIdx.x;
    if (i < n) dst[i] = (_Float16)src[i];
}

// B[K][N] (row-major f32) -> Bt[N][K] (row-major f16): columns become contiguous
__global__ void k_cvt_t_f16(const float* __restrict__ B, _Float16* __restrict__ Bt,
                            int K, int N)
{
    int i = blockIdx.x * blockDim.x + threadIdx.x;   // coalesced read of B
    if (i >= K * N) return;
    int k = i / N;
    int n = i - k * N;
    Bt[(size_t)n * K + k] = (_Float16)B[i];
}

// -----------------------------------------------------------------------------
// C[M,N] = relu(Ah[M,K] @ Bt[N,K]^T + bias[N])  via v_wmma_f32_16x16x32_f16
// Ah: f16 row-major. Bt: f16, B transposed so each output column's K-run is
// contiguous. One wave32 per 16x16 tile; all fragment loads are aligned b128.
// Optionally writes f32 (Cf) and/or f16 (Ch) outputs.
// -----------------------------------------------------------------------------
template <int K>
__global__ __launch_bounds__(256) void k_gemm_relu(
    const _Float16* __restrict__ Ah, const _Float16* __restrict__ Bt,
    const float* __restrict__ bias,
    float* __restrict__ Cf, _Float16* __restrict__ Ch,
    int N, int tilesTotal)
{
    int lane = threadIdx.x & 31;
    int wv   = threadIdx.x >> 5;
    int tile = blockIdx.x * WAVES_PER_BLOCK + wv;
    if (tile >= tilesTotal) return;          // wave-uniform: EXEC stays all-1s

    int tilesN = N >> 4;
    int tm = tile / tilesN;
    int tn = tile - tm * tilesN;

    int l15 = lane & 15;
    int lhi = lane >> 4;

    // A fragment: lane holds row (tm*16+l15); per ISA 16-bit A 16x32 layout the
    // lane's 16 elements are K = {akbase..akbase+7, 16+akbase..16+akbase+7}
    const _Float16* Arow = Ah + (size_t)(tm * 16 + l15) * K;
    const int akbase = lhi * 8;
    // B fragment: lane holds column (tn*16+l15); elements are K = {bkbase..bkbase+7, 16+bkbase..16+bkbase+7}
    const _Float16* Brow = Bt + (size_t)(tn * 16 + l15) * K;
    const int bkbase = lhi * 8;

    v8f acc = {};
#pragma unroll
    for (int k0 = 0; k0 < K; k0 += 32) {
        v8h alo = *(const v8h*)(Arow + k0 + akbase);        // b128
        v8h ahi = *(const v8h*)(Arow + k0 + 16 + akbase);   // b128
        v8h blo = *(const v8h*)(Brow + k0 + bkbase);        // b128
        v8h bhi = *(const v8h*)(Brow + k0 + 16 + bkbase);   // b128
        v16h av = __builtin_shufflevector(alo, ahi, 0, 1, 2, 3, 4, 5, 6, 7,
                                          8, 9, 10, 11, 12, 13, 14, 15);
        v16h bv = __builtin_shufflevector(blo, bhi, 0, 1, 2, 3, 4, 5, 6, 7,
                                          8, 9, 10, 11, 12, 13, 14, 15);
        acc = __builtin_amdgcn_wmma_f32_16x16x32_f16(
            /*neg_a=*/false, av, /*neg_b=*/false, bv,
            /*c_mod=*/(short)0, acc, /*reuse_a=*/false, /*reuse_b=*/false);
    }

    // D layout: col = tn*16 + l15 ; row = tm*16 + lhi*8 + r
    int   col   = tn * 16 + l15;
    float bv0   = bias[col];
    int   rbase = tm * 16 + lhi * 8;
    float vv[8];
#pragma unroll
    for (int r = 0; r < 8; ++r)
        vv[r] = fmaxf(acc[r] + bv0, 0.0f);

    if (Cf) {
#pragma unroll
        for (int r = 0; r < 8; ++r)
            Cf[(size_t)(rbase + r) * N + col] = vv[r];
    }
    if (Ch) {
#pragma unroll
        for (int r = 0; r < 8; ++r)
            Ch[(size_t)(rbase + r) * N + col] = (_Float16)vv[r];
    }
}

// -----------------------------------------------------------------------------
// Per-Gaussian finalize: color head (64->3 + sigmoid) + projection geometry.
// Packs 12 floats per Gaussian: {mx, my, ia, ib, ic, coef, r, g, b, pad...}
// -----------------------------------------------------------------------------
__global__ void k_finalize(
    const float* __restrict__ H2, const float* __restrict__ w3,
    const float* __restrict__ b3,
    const float* __restrict__ pos, const float* __restrict__ scl,
    const float* __restrict__ rot, const float* __restrict__ opa,
    const float* __restrict__ campose,
    float* __restrict__ params, int N, int H2dim)
{
    int n = blockIdx.x * blockDim.x + threadIdx.x;
    if (n >= N) return;
    const float EPS = 1e-6f;

    // ---- color head: sigmoid(H2[n,:] @ w3 + b3)
    float c0 = b3[0], c1 = b3[1], c2 = b3[2];
    const float* h = H2 + (size_t)n * H2dim;
    for (int k = 0; k < H2dim; ++k) {
        float hv = h[k];
        c0 = fmaf(hv, w3[k * 3 + 0], c0);
        c1 = fmaf(hv, w3[k * 3 + 1], c1);
        c2 = fmaf(hv, w3[k * 3 + 2], c2);
    }
    c0 = 1.0f / (1.0f + __expf(-c0));
    c1 = 1.0f / (1.0f + __expf(-c1));
    c2 = 1.0f / (1.0f + __expf(-c2));

    // ---- quaternion -> rotation
    float qw = rot[n * 4 + 0], qx = rot[n * 4 + 1];
    float qy = rot[n * 4 + 2], qz = rot[n * 4 + 3];
    float qn = rsqrtf(qw * qw + qx * qx + qy * qy + qz * qz);
    qw *= qn; qx *= qn; qy *= qn; qz *= qn;
    float R[3][3] = {
        {1.f - 2.f * (qy * qy + qz * qz), 2.f * (qx * qy - qw * qz), 2.f * (qx * qz + qw * qy)},
        {2.f * (qx * qy + qw * qz), 1.f - 2.f * (qx * qx + qz * qz), 2.f * (qy * qz - qw * qx)},
        {2.f * (qx * qz - qw * qy), 2.f * (qy * qz + qw * qx), 1.f - 2.f * (qx * qx + qy * qy)}};

    float s0 = scl[n * 3 + 0], s1 = scl[n * 3 + 1], s2 = scl[n * 3 + 2];
    float d0 = s0 * s0, d1 = s1 * s1, d2 = s2 * s2;
    float M3[3][3];
#pragma unroll
    for (int i = 0; i < 3; ++i)
#pragma unroll
        for (int k = 0; k < 3; ++k)
            M3[i][k] = R[i][0] * d0 * R[k][0] + R[i][1] * d1 * R[k][1] + R[i][2] * d2 * R[k][2];

    // ---- camera transform (pc = Rc * pos + t), camera_pose is row-major 4x4
    float p0 = pos[n * 3 + 0], p1 = pos[n * 3 + 1], p2 = pos[n * 3 + 2];
    float pc[3];
#pragma unroll
    for (int i = 0; i < 3; ++i)
        pc[i] = campose[i * 4 + 0] * p0 + campose[i * 4 + 1] * p1 +
                campose[i * 4 + 2] * p2 + campose[i * 4 + 3];

    float iz = 1.0f / pc[2];
    float mx = pc[0] * iz, my = pc[1] * iz;
    float J0[3] = {iz, 0.f, -pc[0] * iz * iz};
    float J1[3] = {0.f, iz, -pc[1] * iz * iz};

    float MJ0[3], MJ1[3];
#pragma unroll
    for (int i = 0; i < 3; ++i) {
        MJ0[i] = M3[i][0] * J0[0] + M3[i][1] * J0[1] + M3[i][2] * J0[2];
        MJ1[i] = M3[i][0] * J1[0] + M3[i][1] * J1[1] + M3[i][2] * J1[2];
    }
    float c00 = J0[0] * MJ0[0] + J0[1] * MJ0[1] + J0[2] * MJ0[2];
    float c01 = J0[0] * MJ1[0] + J0[1] * MJ1[1] + J0[2] * MJ1[2];
    float c11 = J1[0] * MJ1[0] + J1[1] * MJ1[1] + J1[2] * MJ1[2];

    float a = c00 + EPS, b = c01, c = c11 + EPS;
    float idet = 1.0f / (a * c - b * b);
    float ia = c * idet, ib = -b * idet, ic = a * idet;
    float det0 = c00 * c11 - c01 * c01;
    float coef = opa[n] / (6.28318530717958647692f * sqrtf(det0 + EPS));

    float* pp = params + (size_t)n * 12;
    pp[0] = mx; pp[1] = my; pp[2] = ia; pp[3] = ib; pp[4] = ic;
    pp[5] = coef; pp[6] = c0; pp[7] = c1; pp[8] = c2;
    pp[9] = 0.f; pp[10] = 0.f; pp[11] = 0.f;
}

// -----------------------------------------------------------------------------
// Render: one thread per pixel, sequential front-to-back alpha compositing
// (matches the reference cumprod). Gaussian params staged in LDS (48 KB of the
// 320 KB WGP LDS) and broadcast-read by all 8 waves.
// -----------------------------------------------------------------------------
__global__ __launch_bounds__(256) void k_render(
    const float* __restrict__ params, float* __restrict__ out,
    int N, const int* __restrict__ dW, int P)
{
    __shared__ float sg[MAXN * 12];
    int tid = threadIdx.x;
    int total = N * 12;
    for (int i = tid; i < total; i += blockDim.x) sg[i] = params[i];
    __syncthreads();

    int pix = blockIdx.x * blockDim.x + tid;
    if (pix >= P) return;

    int W = dW[0];
    float px = (float)(pix % W);
    float py = (float)(pix / W);

    float T = 1.0f, r = 0.0f, g = 0.0f, b = 0.0f;
    for (int n = 0; n < N; ++n) {
        const float* s = sg + n * 12;
        float dx = px - s[0];
        float dy = py - s[1];
        float q = s[2] * dx * dx + 2.0f * s[3] * dx * dy + s[4] * dy * dy;
        float alpha = s[5] * __expf(-0.5f * q);
        float w = alpha * T;
        r = fmaf(w, s[6], r);
        g = fmaf(w, s[7], g);
        b = fmaf(w, s[8], b);
        T = T * (1.0f - alpha);
    }
    float* o = out + (size_t)pix * 3;
    o[0] = r; o[1] = g; o[2] = b;
}

// -----------------------------------------------------------------------------
extern "C" void kernel_launch(void* const* d_in, const int* in_sizes, int n_in,
                              void* d_out, int out_size, void* d_ws, size_t ws_size,
                              hipStream_t stream)
{
    const float* positions = (const float*)d_in[0];
    const float* scales    = (const float*)d_in[1];
    const float* rotations = (const float*)d_in[2];
    const float* opacities = (const float*)d_in[3];
    const float* features  = (const float*)d_in[4];
    // d_in[5] camera_matrix: unused by the reference math
    const float* campose   = (const float*)d_in[6];
    const float* w1 = (const float*)d_in[7];
    const float* b1 = (const float*)d_in[8];
    const float* w2 = (const float*)d_in[9];
    const float* b2 = (const float*)d_in[10];
    const float* w3 = (const float*)d_in[11];
    const float* b3 = (const float*)d_in[12];
    const int*   dW = (const int*)d_in[14];
    (void)n_in; (void)ws_size;

    const int M     = in_sizes[0] / 3;   // 1024 gaussians
    const int FD    = in_sizes[4] / M;   // 256
    const int H1dim = in_sizes[8];       // 128
    const int H2dim = in_sizes[10];      // 64
    const int P     = out_size / 3;      // H*W pixels

    // Workspace layout (all 16B-aligned offsets):
    _Float16* A1h = (_Float16*)d_ws;                       // M * FD
    _Float16* B1t = A1h + (size_t)M * FD;                  // H1dim * FD
    _Float16* B2t = B1t + (size_t)H1dim * FD;              // H2dim * H1dim
    _Float16* H1h = B2t + (size_t)H2dim * H1dim;           // M * H1dim
    float*    H2  = (float*)(H1h + (size_t)M * H1dim);     // M * H2dim
    float* params = H2 + (size_t)M * H2dim;                // M * 12

    // Layout prep: f16 convert A, f16 convert+transpose weights
    {
        int n = M * FD;
        k_cvt_f16<<<(n + 255) / 256, 256, 0, stream>>>(features, A1h, n);
        int n1 = FD * H1dim;
        k_cvt_t_f16<<<(n1 + 255) / 256, 256, 0, stream>>>(w1, B1t, FD, H1dim);
        int n2 = H1dim * H2dim;
        k_cvt_t_f16<<<(n2 + 255) / 256, 256, 0, stream>>>(w2, B2t, H1dim, H2dim);
    }

    // Layer 1: relu(features @ w1 + b1) -> f16 H1h   (WMMA f16->f32)
    {
        int tiles = (M / 16) * (H1dim / 16);
        int blocks = (tiles + WAVES_PER_BLOCK - 1) / WAVES_PER_BLOCK;
        if (FD == 256)
            k_gemm_relu<256><<<blocks, 256, 0, stream>>>(A1h, B1t, b1, nullptr, H1h, H1dim, tiles);
        else
            k_gemm_relu<128><<<blocks, 256, 0, stream>>>(A1h, B1t, b1, nullptr, H1h, H1dim, tiles);
    }
    // Layer 2: relu(H1 @ w2 + b2) -> f32 H2   (WMMA f16->f32)
    {
        int tiles = (M / 16) * (H2dim / 16);
        int blocks = (tiles + WAVES_PER_BLOCK - 1) / WAVES_PER_BLOCK;
        k_gemm_relu<128><<<blocks, 256, 0, stream>>>(H1h, B2t, b2, H2, nullptr, H2dim, tiles);
    }
    // Color head + geometry -> packed per-gaussian params
    k_finalize<<<(M + 255) / 256, 256, 0, stream>>>(
        H2, w3, b3, positions, scales, rotations, opacities, campose,
        params, M, H2dim);

    // Alpha compositing over all pixels
    k_render<<<(P + 255) / 256, 256, 0, stream>>>(params, (float*)d_out, M, dW, P);
}
